// MultiheadSelfAttention_82918638616721
// MI455X (gfx1250) — matrix operand order repaired and checked by reading, not compiled, not run on hardware
//
#include <hip/hip_runtime.h>
#include <hip/hip_bf16.h>

typedef __attribute__((ext_vector_type(16))) _Float16 v16h;
typedef __attribute__((ext_vector_type(8)))  _Float16 v8h;
typedef __attribute__((ext_vector_type(8)))  float    v8f;

#define B_  2
#define S_  2048
#define E_  1024
#define H_  16
#define D_  64
#define R_  (B_ * S_)   // 4096 token rows
#define F3_ (3 * E_)    // 3072 qkv features

// ---------------------------------------------------------------------------
// WMMA fragment helpers (CDNA5 wave32 layouts, CDNA5 ISA §7.12.2)
// ---------------------------------------------------------------------------

// A-matrix f16 16x32 (MxK), row-major source with leading dim `lda` (elements).
// lane = M + 16*hi ; element j -> K = (j&7) + 8*hi + 16*(j>>3)
// => two contiguous 8-halfword (16B) runs at K = 8*hi and K = 16+8*hi.
static __device__ __forceinline__ v16h load_a16(const _Float16* __restrict__ A, int lda) {
    const int lane = threadIdx.x & 31;
    const int m = lane & 15, hi = lane >> 4;
    const _Float16* p = A + m * lda + 8 * hi;
    v8h lo = *(const v8h*)(p);
    v8h hh = *(const v8h*)(p + 16);
    v16h a;
#pragma unroll
    for (int j = 0; j < 8; ++j) { a[j] = lo[j]; a[j + 8] = hh[j]; }
    return a;
}

// B-matrix f16 32x16 (KxN) from a transposed [N][K] buffer (leading dim ldk):
// lane = N + 16*khi ; element j -> K = (j&7) + 8*khi + 16*(j>>3)  => two 8-halfword runs.
static __device__ __forceinline__ v16h load_b16(const _Float16* __restrict__ Bt, int ldk) {
    const int lane = threadIdx.x & 31;
    const int n = lane & 15, khi = lane >> 4;
    const _Float16* p = Bt + n * ldk + 8 * khi;
    v8h lo = *(const v8h*)(p);
    v8h hh = *(const v8h*)(p + 16);
    v16h b;
#pragma unroll
    for (int j = 0; j < 8; ++j) { b[j] = lo[j]; b[j + 8] = hh[j]; }
    return b;
}

static __device__ __forceinline__ v8f wmma_f16(v16h a, v16h b, v8f c) {
    return __builtin_amdgcn_wmma_f32_16x16x32_f16(
        /*neg_a=*/false, a, /*neg_b=*/false, b,
        /*c_mod=*/(short)0, c, /*reuse_a=*/false, /*reuse_b=*/false);
}

static __device__ __forceinline__ v8f vzero8() {
    v8f z = {0.f, 0.f, 0.f, 0.f, 0.f, 0.f, 0.f, 0.f};
    return z;
}

// 16-byte async global->LDS copy (CDNA5 GLOBAL_LOAD_ASYNC_TO_LDS_B128,
// tracked by ASYNCcnt; CDNA5 ISA).
static __device__ __forceinline__ void async_copy16(_Float16* lds_dst,
                                                    const _Float16* gsrc) {
    const unsigned loff = (unsigned)(unsigned long long)lds_dst;  // low 32b = LDS offset
    asm volatile("global_load_async_to_lds_b128 %0, %1, off"
                 :: "v"(loff), "v"(gsrc) : "memory");
}

// ---------------------------------------------------------------------------
// Precision conversion / layout kernels
// ---------------------------------------------------------------------------

__global__ void mha_cvt_f16_kernel(const float* __restrict__ src,
                                   _Float16* __restrict__ dst, int n) {
    int i = blockIdx.x * blockDim.x + threadIdx.x;
    if (i < n) dst[i] = (_Float16)src[i];
}

// dst[c*rows + r] = (f16) src[r*cols + c]  (weights -> [out_feat][in_feat])
__global__ void mha_transpose_cvt_kernel(const float* __restrict__ src,
                                         _Float16* __restrict__ dst,
                                         int rows, int cols) {
    int i = blockIdx.x * blockDim.x + threadIdx.x;
    if (i < rows * cols) {
        int r = i / cols, c = i - r * cols;
        dst[c * rows + r] = (_Float16)src[i];
    }
}

// ---------------------------------------------------------------------------
// GEMM1: QKV projection.  Xh[4096x1024] * WqkvT[3072x1024]^T
// One wave -> 16x64 output strip.  Ping-pong double buffering over 64-wide
// K steps: loads write directly into the set consumed next (no reg copies),
// WMMAs overlap the in-flight loads via partial s_wait_loadcnt.
// Scatter into Q/K (row-major per head) and V transposed [b][h][d][s].
// ---------------------------------------------------------------------------
__global__ void __launch_bounds__(256)
mha_qkv_gemm_kernel(const _Float16* __restrict__ Xh,
                    const _Float16* __restrict__ WqkvT,
                    _Float16* __restrict__ Qh,
                    _Float16* __restrict__ Kh,
                    _Float16* __restrict__ Vt) {
    const int NSTRIPS = F3_ / 64;  // 48
    const int wid = blockIdx.x * 8 + (threadIdx.x >> 5);
    const int mtile = wid / NSTRIPS;
    const int nstrip = wid - mtile * NSTRIPS;
    const int lane = threadIdx.x & 31;
    const int n = lane & 15, hi = lane >> 4;

    v8f c[4];
#pragma unroll
    for (int j = 0; j < 4; ++j) c[j] = vzero8();

    const _Float16* Arow = Xh + (size_t)mtile * 16 * E_;
    const _Float16* Brow = WqkvT + (size_t)(nstrip * 64) * E_;

    v16h a0, a1, b0[4], b1[4];
    a0 = load_a16(Arow, E_);
#pragma unroll
    for (int j = 0; j < 4; ++j) b0[j] = load_b16(Brow + (size_t)(j * 16) * E_, E_);

    for (int k0 = 0; k0 < E_; k0 += 64) {
        // prefetch k0+32 into set1 (always in range: E % 64 == 0)
        a1 = load_a16(Arow + k0 + 32, E_);
#pragma unroll
        for (int j = 0; j < 4; ++j)
            b1[j] = load_b16(Brow + (size_t)(j * 16) * E_ + k0 + 32, E_);
        // compute k0 from set0
#pragma unroll
        for (int j = 0; j < 4; ++j) c[j] = wmma_f16(a0, b0[j], c[j]);
        // prefetch k0+64 into set0 (skipped only on final step)
        if (k0 + 64 < E_) {
            a0 = load_a16(Arow + k0 + 64, E_);
#pragma unroll
            for (int j = 0; j < 4; ++j)
                b0[j] = load_b16(Brow + (size_t)(j * 16) * E_ + k0 + 64, E_);
        }
        // compute k0+32 from set1
#pragma unroll
        for (int j = 0; j < 4; ++j) c[j] = wmma_f16(a1, b1[j], c[j]);
    }

    const int sec = (nstrip * 64) >> 10;  // 0=Q 1=K 2=V (strips never straddle)
#pragma unroll
    for (int j = 0; j < 4; ++j) {
        const int f = (nstrip * 64 + j * 16 + n) & (E_ - 1);
        const int h = f >> 6, d = f & 63;
#pragma unroll
        for (int r = 0; r < 8; ++r) {
            const int row = mtile * 16 + r + 8 * hi;    // 0..4095
            const int bb = row >> 11, s = row & (S_ - 1);
            const _Float16 val = (_Float16)c[j][r];
            if (sec == 0)      Qh[(((size_t)(bb * H_ + h) * S_ + s) * D_) + d] = val;
            else if (sec == 1) Kh[(((size_t)(bb * H_ + h) * S_ + s) * D_) + d] = val;
            else               Vt[(((size_t)(bb * H_ + h) * D_ + d) * S_) + s] = val;
        }
    }
}

// ---------------------------------------------------------------------------
// Flash attention.  4 waves/block share one (b,h); each wave owns 16 query
// rows.  K/V tiles for each 32-key step are staged cooperatively into LDS by
// the async DMA engine (GLOBAL_LOAD_ASYNC_TO_LDS_B128, double-buffered with
// s_wait_asynccnt), then consumed as WMMA B fragments via ds_load.
// ---------------------------------------------------------------------------
__global__ void __launch_bounds__(128)
mha_flash_attn_kernel(const _Float16* __restrict__ Qh,
                      const _Float16* __restrict__ Kh,
                      const _Float16* __restrict__ Vt,
                      _Float16* __restrict__ Oh) {
    __shared__ _Float16 Ktile[2][32 * 64];  // [key t (32)][d (64)]      4KB x2
    __shared__ _Float16 Vtile[2][64 * 32];  // [d (64)][key t (32)]      4KB x2
    __shared__ _Float16 pbuf[4][16 * 32];   // per-wave P staging        4KB

    const int tid  = threadIdx.x;
    const int wave = tid >> 5;
    const int lane = tid & 31;
    const int n = lane & 15, hi = lane >> 4;
    const int bh = blockIdx.y;            // 0..B*H-1
    const int b = bh >> 4, h = bh & (H_ - 1);
    const int m0 = (blockIdx.x * 4 + wave) * 16;

    const _Float16* Qbase = Qh + ((size_t)bh * S_ + m0) * D_;
    const _Float16* Kbase = Kh + (size_t)bh * S_ * D_;
    const _Float16* Vbase = Vt + (size_t)bh * D_ * S_;

    // Stage one 32-key step: K rows t0..t0+31 (32x128B) + V rows d=0..63
    // (64x64B).  512 x 16B chunks, 4 per thread -> 4 async ops in flight.
    auto stage = [&](int buf, int t0) {
#pragma unroll
        for (int i = 0; i < 2; ++i) {   // K tile: chunks 0..255
            const int ck = tid + 128 * i;
            const int row = ck >> 3, off = (ck & 7) * 8;
            async_copy16(&Ktile[buf][row * 64 + off],
                         Kbase + (size_t)(t0 + row) * D_ + off);
        }
#pragma unroll
        for (int i = 0; i < 2; ++i) {   // V tile: chunks 0..255
            const int cv = tid + 128 * i;
            const int d = cv >> 2, off = (cv & 3) * 8;
            async_copy16(&Vtile[buf][d * 32 + off],
                         Vbase + (size_t)d * S_ + t0 + off);
        }
    };

    // Q A-fragments for d = 0..31 and 32..63, resident for the whole pass.
    v16h qa0 = load_a16(Qbase + 0, D_);
    v16h qa1 = load_a16(Qbase + 32, D_);

    v8f o[4];
#pragma unroll
    for (int j = 0; j < 4; ++j) o[j] = vzero8();
    float rmax[8], rsum[8], corr[8];
#pragma unroll
    for (int r = 0; r < 8; ++r) { rmax[r] = -1e30f; rsum[r] = 0.f; }

    const float scale = 0.125f;  // 1/sqrt(64)

    stage(0, 0);
    int buf = 0;
    for (int t0 = 0; t0 < S_; t0 += 32, buf ^= 1) {
        if (t0 + 32 < S_) {
            stage(buf ^ 1, t0 + 32);  // 4 new async ops outstanding
            asm volatile("s_wait_asynccnt 0x4" ::: "memory");  // drain current buf only
        } else {
            asm volatile("s_wait_asynccnt 0x0" ::: "memory");
        }
        __syncthreads();  // current K/V tile visible to all waves

        const _Float16* Kt = &Ktile[buf][0];
        const _Float16* Vti = &Vtile[buf][0];

        // ---- scores: two 16x16 tiles (keys t0..+15 and t0+16..+31) ----
        v8f s0 = vzero8(), s1 = vzero8();
        {
            v16h kb0 = load_b16(Kt + 0 * 64 + 0, 64);
            v16h kb1 = load_b16(Kt + 0 * 64 + 32, 64);
            s0 = wmma_f16(qa0, kb0, s0);
            s0 = wmma_f16(qa1, kb1, s0);
        }
        {
            v16h kb0 = load_b16(Kt + 16 * 64 + 0, 64);
            v16h kb1 = load_b16(Kt + 16 * 64 + 32, 64);
            s1 = wmma_f16(qa0, kb0, s1);
            s1 = wmma_f16(qa1, kb1, s1);
        }

        // ---- online softmax update (rows r+8*hi live across lanes n=0..15) --
#pragma unroll
        for (int r = 0; r < 8; ++r) {
            float v0 = s0[r] * scale, v1 = s1[r] * scale;
            float mx = fmaxf(v0, v1);
            mx = fmaxf(mx, __shfl_xor(mx, 1));
            mx = fmaxf(mx, __shfl_xor(mx, 2));
            mx = fmaxf(mx, __shfl_xor(mx, 4));
            mx = fmaxf(mx, __shfl_xor(mx, 8));
            const float nm = fmaxf(rmax[r], mx);
            corr[r] = __expf(rmax[r] - nm);
            rmax[r] = nm;
            const float p0 = __expf(v0 - nm);
            const float p1 = __expf(v1 - nm);
            s0[r] = p0; s1[r] = p1;
            float ps = p0 + p1;
            ps += __shfl_xor(ps, 1);
            ps += __shfl_xor(ps, 2);
            ps += __shfl_xor(ps, 4);
            ps += __shfl_xor(ps, 8);
            rsum[r] = rsum[r] * corr[r] + ps;
        }

        // ---- P: C-layout f32 -> LDS -> A-layout f16 (per-wave buffer) ----
        _Float16* pb = &pbuf[wave][0];
#pragma unroll
        for (int r = 0; r < 8; ++r) {
            const int row = r + 8 * hi;
            pb[row * 32 + n]      = (_Float16)s0[r];
            pb[row * 32 + 16 + n] = (_Float16)s1[r];
        }
        v16h pa;
        {
            // same-wave LDS ops are in-order; read back in A layout
            const _Float16* p = pb + (lane & 15) * 32 + 8 * hi;
            v8h lo = *(const v8h*)(p);
            v8h hh = *(const v8h*)(p + 16);
#pragma unroll
            for (int j = 0; j < 8; ++j) { pa[j] = lo[j]; pa[j + 8] = hh[j]; }
        }

        // ---- O = O*corr + P @ V  (V tile transposed: contiguous along t) ----
#pragma unroll
        for (int dt = 0; dt < 4; ++dt) {
#pragma unroll
            for (int r = 0; r < 8; ++r) o[dt][r] *= corr[r];
            v16h vb = load_b16(Vti + (dt * 16) * 32, 32);
            o[dt] = wmma_f16(pa, vb, o[dt]);
        }

        __syncthreads();  // all waves done reading this buf before DMA reuses it
    }

    // ---- epilogue: normalize, write Oh[b*S+s][h*64+d] as f16 ----
#pragma unroll
    for (int dt = 0; dt < 4; ++dt) {
#pragma unroll
        for (int r = 0; r < 8; ++r) {
            const float v = o[dt][r] / rsum[r];
            const int s = m0 + r + 8 * hi;
            Oh[((size_t)(b * S_ + s)) * E_ + h * D_ + dt * 16 + n] = (_Float16)v;
        }
    }
}

// ---------------------------------------------------------------------------
// GEMM2: out = Oh[4096x1024] * WoutT[1024x1024]^T -> fp32 d_out
// Same ping-pong pipelined structure as GEMM1.
// ---------------------------------------------------------------------------
__global__ void __launch_bounds__(256)
mha_out_gemm_kernel(const _Float16* __restrict__ Oh,
                    const _Float16* __restrict__ WoutT,
                    float* __restrict__ out) {
    const int NSTRIPS = E_ / 64;  // 16
    const int wid = blockIdx.x * 8 + (threadIdx.x >> 5);
    const int mtile = wid / NSTRIPS;
    const int nstrip = wid - mtile * NSTRIPS;
    const int lane = threadIdx.x & 31;
    const int n = lane & 15, hi = lane >> 4;

    v8f c[4];
#pragma unroll
    for (int j = 0; j < 4; ++j) c[j] = vzero8();

    const _Float16* Arow = Oh + (size_t)mtile * 16 * E_;
    const _Float16* Brow = WoutT + (size_t)(nstrip * 64) * E_;

    v16h a0, a1, b0[4], b1[4];
    a0 = load_a16(Arow, E_);
#pragma unroll
    for (int j = 0; j < 4; ++j) b0[j] = load_b16(Brow + (size_t)(j * 16) * E_, E_);

    for (int k0 = 0; k0 < E_; k0 += 64) {
        a1 = load_a16(Arow + k0 + 32, E_);
#pragma unroll
        for (int j = 0; j < 4; ++j)
            b1[j] = load_b16(Brow + (size_t)(j * 16) * E_ + k0 + 32, E_);
#pragma unroll
        for (int j = 0; j < 4; ++j) c[j] = wmma_f16(a0, b0[j], c[j]);
        if (k0 + 64 < E_) {
            a0 = load_a16(Arow + k0 + 64, E_);
#pragma unroll
            for (int j = 0; j < 4; ++j)
                b0[j] = load_b16(Brow + (size_t)(j * 16) * E_ + k0 + 64, E_);
        }
#pragma unroll
        for (int j = 0; j < 4; ++j) c[j] = wmma_f16(a1, b1[j], c[j]);
    }

#pragma unroll
    for (int j = 0; j < 4; ++j) {
        const int col = nstrip * 64 + j * 16 + n;
#pragma unroll
        for (int r = 0; r < 8; ++r) {
            const int row = mtile * 16 + r + 8 * hi;
            out[(size_t)row * E_ + col] = c[j][r];
        }
    }
}

// ---------------------------------------------------------------------------
// Host launcher
// ---------------------------------------------------------------------------
extern "C" void kernel_launch(void* const* d_in, const int* in_sizes, int n_in,
                              void* d_out, int out_size, void* d_ws, size_t ws_size,
                              hipStream_t stream) {
    const float* x     = (const float*)d_in[0];  // (B,S,E)
    const float* w_qkv = (const float*)d_in[1];  // (E,3E)
    const float* w_out = (const float*)d_in[2];  // (E,E)

    char* ws = (char*)d_ws;
    _Float16* Xh    = (_Float16*)(ws + (size_t)0);          //  8 MiB
    _Float16* WqkvT = (_Float16*)(ws + ((size_t)8  << 20)); //  6 MiB
    _Float16* WoutT = (_Float16*)(ws + ((size_t)14 << 20)); //  2 MiB
    _Float16* Qh    = (_Float16*)(ws + ((size_t)16 << 20)); //  8 MiB
    _Float16* Kh    = (_Float16*)(ws + ((size_t)24 << 20)); //  8 MiB
    _Float16* Vt    = (_Float16*)(ws + ((size_t)32 << 20)); //  8 MiB  (transposed V)
    _Float16* Oh    = (_Float16*)(ws + ((size_t)40 << 20)); //  8 MiB

    // 1) precision conversion + weight transposition
    mha_cvt_f16_kernel<<<(R_ * E_ + 255) / 256, 256, 0, stream>>>(x, Xh, R_ * E_);
    mha_transpose_cvt_kernel<<<(E_ * F3_ + 255) / 256, 256, 0, stream>>>(w_qkv, WqkvT, E_, F3_);
    mha_transpose_cvt_kernel<<<(E_ * E_ + 255) / 256, 256, 0, stream>>>(w_out, WoutT, E_, E_);

    // 2) QKV projection (WMMA, ping-pong pipelined): 256 M-tiles x 48 N-strips
    mha_qkv_gemm_kernel<<<(R_ / 16) * (F3_ / 64) / 8, 256, 0, stream>>>(Xh, WqkvT, Qh, Kh, Vt);

    // 3) flash attention (WMMA + async-to-LDS K/V staging + online softmax)
    mha_flash_attn_kernel<<<dim3(S_ / 64, B_ * H_), 128, 0, stream>>>(Qh, Kh, Vt, Oh);

    // 4) output projection (WMMA, ping-pong pipelined), fp32 store
    mha_out_gemm_kernel<<<(R_ / 16) * (E_ / 64) / 8, 256, 0, stream>>>(Oh, WoutT, (float*)d_out);
}
